// HNC_strategy_37512244363642
// MI455X (gfx1250) — matrix operand order repaired and checked by reading, not compiled, not run on hardware
//
#include <hip/hip_runtime.h>
#include <math.h>

// ---------------------------------------------------------------------------
// Problem constants (from reference)
// ---------------------------------------------------------------------------
#define BN    16384      // batch
#define DIMC  64
#define MLPS  65
#define HC    66
#define O1    8320       // MLPS*2*DIM
#define O2    8385       // O1 + MLPS
#define O3    12545      // O2 + DIM*MLPS
// padded shapes
#define KT    96         // trunk K (65/66 real + bias + zero pad), 3 wmma k-steps
#define NT    80         // trunk hidden N pad (66 -> 80), 5 n-tiles
#define KH    160        // hyper GEMM1 K (129 real -> 160), 5 k-steps
#define N1    1040       // 65 m-tiles * 16 packed k-channels
#define K2    96         // hyper GEMM2 K (65 real -> 96), 3 k-steps
#define N2    1024       // 64 o-tiles * 16 packed k-channels
#define LDSS  112        // LDS row stride in f16 (224B, multiple of 32B)

typedef __attribute__((ext_vector_type(16))) _Float16 v16h;
typedef __attribute__((ext_vector_type(8)))  _Float16 v8h;
typedef __attribute__((ext_vector_type(8)))  float    v8f;

// ---------------------------------------------------------------------------
// Fast helpers
// ---------------------------------------------------------------------------
// Branch-free tanh: 1 - 2/(1 + 2^(2*log2(e)*x)).  Lowers to v_exp_f32 +
// v_rcp_f32 + 2 VALU; saturates correctly to +/-1 at both extremes.
__device__ __forceinline__ float fast_tanh(float x) {
  const float e = __builtin_amdgcn_exp2f(x * 2.88539008177792681472f); // 2^(2x*log2e)
  const float r = __builtin_amdgcn_rcpf(1.0f + e);
  return 1.0f - 2.0f * r;
}

// Single-instruction XOR lane shuffle via ds_swizzle group-of-32 mode:
// offset = {1'b0, xor_mask[4:0]=MASK, or_mask=0, and_mask=0x1f}.
// For MASK < 16 this never crosses a 16-lane half, exactly what the
// packed-k reduction needs.
template <int MASK>
__device__ __forceinline__ float xor_swizzle(float v) {
  int i = __builtin_bit_cast(int, v);
  i = __builtin_amdgcn_ds_swizzle(i, (MASK << 10) | 0x1f);
  return __builtin_bit_cast(float, i);
}

__device__ __forceinline__ void butterfly16(float (&red)[8]) {
#pragma unroll
  for (int j = 0; j < 8; ++j) red[j] += xor_swizzle<1>(red[j]);
#pragma unroll
  for (int j = 0; j < 8; ++j) red[j] += xor_swizzle<2>(red[j]);
#pragma unroll
  for (int j = 0; j < 8; ++j) red[j] += xor_swizzle<4>(red[j]);
#pragma unroll
  for (int j = 0; j < 8; ++j) red[j] += xor_swizzle<8>(red[j]);
}

// ---------------------------------------------------------------------------
// WMMA helpers (CDNA5: wave32, D = A(16x32 f16) * B(32x16 f16) + C(16x16 f32))
// ---------------------------------------------------------------------------
__device__ __forceinline__ v8f wmma16(v16h a, v16h b, v8f c) {
  return __builtin_amdgcn_wmma_f32_16x16x32_f16(
      /*neg_a=*/false, a, /*neg_b=*/false, b,
      /*c_mod=*/(short)0, c, /*reuse_a=*/false, /*reuse_b=*/false);
}

// A-matrix fragment, 16x32 f16, source row-major [16][stride].
// ISA layout: lanes 0-15 -> M=lane, K={0..7} in v0-3 and {16..23} in v4-7;
//             lanes16-31 -> M=lane-16, K={8..15} and {24..31}.
__device__ __forceinline__ v16h load_a_frag(const _Float16* base, int stride,
                                            int kstep, int lane) {
  const int r  = lane & 15;
  const int kb = kstep * 32 + ((lane >> 4) << 3);       // +0 or +8
  const _Float16* p = base + r * stride + kb;
  v8h lo = *(const v8h*)(p);
  v8h hi = *(const v8h*)(p + 16);
  return __builtin_shufflevector(lo, hi, 0,1,2,3,4,5,6,7,8,9,10,11,12,13,14,15);
}

// B-matrix fragment, 32x16 f16, weights stored TRANSPOSED row-major [N][stride=Kpad]
// so each lane reads two 16B runs, 32B apart.
// ISA layout: lanes 0-15 -> N=lane, K={0..7,16..23}; lanes 16-31 -> N=lane-16, K={8..15,24..31}.
__device__ __forceinline__ v16h load_b_frag(const _Float16* base, int stride,
                                            int kstep, int lane) {
  const int n  = lane & 15;
  const int kb = kstep * 32 + ((lane >> 4) << 3);       // +0 or +8
  const _Float16* p = base + n * stride + kb;
  v8h lo = *(const v8h*)(p);
  v8h hi = *(const v8h*)(p + 16);
  return __builtin_shufflevector(lo, hi, 0,1,2,3,4,5,6,7,8,9,10,11,12,13,14,15);
}

// ---------------------------------------------------------------------------
// Prep: inputs  Z (B x 96 f16) = [t-0.5, freqs, 1, 0...]
//               XH (B x 160 f16) = [cos(y0..63), sin(y0..63), 1, 0...]
// ---------------------------------------------------------------------------
__global__ void prep_inputs_kernel(const float* __restrict__ t,
                                   const float* __restrict__ y,
                                   const float* __restrict__ freqs,
                                   _Float16* __restrict__ Z,
                                   _Float16* __restrict__ XH) {
  const int i = blockIdx.x * blockDim.x + threadIdx.x;
  if (i < BN * KH) {
    const int b = i / KH, c = i % KH;
    float v = 0.f;
    if (c < 64)        v = __cosf(y[b * 128 + c]);
    else if (c < 128)  v = __sinf(y[b * 128 + (c - 64)]);
    else if (c == 128) v = 1.f;
    XH[i] = (_Float16)v;
  }
  if (i < BN * KT) {
    const int b = i / KT, c = i % KT;
    float v = 0.f;
    if (c == 0)        v = t[0] - 0.5f;
    else if (c < 65)   v = freqs[b * 64 + (c - 1)];
    else if (c == 65)  v = 1.f;
    Z[i] = (_Float16)v;
  }
}

// ---------------------------------------------------------------------------
// Prep: trunk weights, stored transposed [N][K] with bias folded at K index
// ---------------------------------------------------------------------------
__global__ void build_trunk_weights_kernel(
    const float* __restrict__ hw0, const float* __restrict__ hb0,
    const float* __restrict__ hw1, const float* __restrict__ hb1,
    const float* __restrict__ hw2, const float* __restrict__ hb2,
    const float* __restrict__ hw3, const float* __restrict__ hb3,
    _Float16* __restrict__ W0, _Float16* __restrict__ W1,
    _Float16* __restrict__ W2, _Float16* __restrict__ W3) {
  const int tid = blockIdx.x * blockDim.x + threadIdx.x;
  if (tid >= NT * KT) return;
  const int j = tid / KT, i = tid % KT;
  float v0 = 0.f, v1 = 0.f, v2 = 0.f;
  if (j < HC) {
    if (i < 65)       v0 = hw0[j * 65 + i];   // hw0 is (66,65) row-major
    else if (i == 65) v0 = hb0[j];            // z bias channel at K=65
    if (i < 66)       { v1 = hw1[j * 66 + i]; v2 = hw2[j * 66 + i]; }
    else if (i == 66) { v1 = hb1[j]; v2 = hb2[j]; }   // h bias channel at K=66
  }
  W0[tid] = (_Float16)v0; W1[tid] = (_Float16)v1; W2[tid] = (_Float16)v2;
  if (tid < 16 * KT) {
    const int j3 = tid / KT, i3 = tid % KT;
    float v3 = 0.f;
    if (j3 < 8) {
      if (i3 < 66)       v3 = hw3[j3 * 66 + i3];
      else if (i3 == 66) v3 = hb3[j3];
    }
    W3[tid] = (_Float16)v3;
  }
}

// ---------------------------------------------------------------------------
// Prep: packed hyper-head weights.
//   A1[n=m*16+kk][i<160]: k-channel kk of W1-hat (i=128 row carries b1; kk=8 carries head_b)
//   A2[n=o*16+kk][m<96] : k-channel kk of W2-hat (kk=8 carries head_b)
//   B2W[o*16+kk]        : b2's k-channels (f32)
// ---------------------------------------------------------------------------
__global__ void build_head_weights_kernel(const float* __restrict__ head_w,
                                          const float* __restrict__ head_b,
                                          _Float16* __restrict__ A1,
                                          _Float16* __restrict__ A2,
                                          float* __restrict__ B2W) {
  const int tid = blockIdx.x * blockDim.x + threadIdx.x;
  if (tid < N1 * KH) {
    const int n = tid / KH, i = tid % KH;
    const int m = n >> 4, kk = n & 15;
    float v = 0.f;
    if (kk < 9 && i < 129) {
      const int r = (i < 128) ? (m * 128 + i) : (O1 + m);
      v = (kk < 8) ? head_w[r * 8 + kk] : head_b[r];
    }
    A1[tid] = (_Float16)v;
  }
  if (tid < N2 * K2) {
    const int n = tid / K2, mi = tid % K2;
    const int o = n >> 4, kk = n & 15;
    float v = 0.f;
    if (kk < 9 && mi < 65) {
      const int r = O2 + o * 65 + mi;
      v = (kk < 8) ? head_w[r * 8 + kk] : head_b[r];
    }
    A2[tid] = (_Float16)v;
  }
  if (tid < 1024) {
    const int o = tid >> 4, kk = tid & 15;
    float v = 0.f;
    const int r = O3 + o;
    if (kk < 8)       v = head_w[r * 8 + kk];
    else if (kk == 8) v = head_b[r];
    B2W[tid] = v;
  }
}

// ---------------------------------------------------------------------------
// Trunk MLP: 4 chained WMMA layers, one 16-row tile per wave, LDS ping-pong.
// Output HT (B x 16 f32) = [tanh(h0..h7), 1, 0..0]  (the h-tilde channels)
// ---------------------------------------------------------------------------
__global__ __launch_bounds__(256) void trunk_kernel(
    const _Float16* __restrict__ Z,
    const _Float16* __restrict__ W0, const _Float16* __restrict__ W1,
    const _Float16* __restrict__ W2, const _Float16* __restrict__ W3,
    float* __restrict__ HT) {
  __shared__ __align__(32) _Float16 sbuf[8][2][16][LDSS];
  const int lane = threadIdx.x & 31;
  const int wave = threadIdx.x >> 5;
  const int half = lane >> 4, ln = lane & 15;
  const int rowBase = (blockIdx.x * 8 + wave) * 16;

  // zero both ping-pong tiles (padding columns must be clean zeros)
  {
    _Float16* p = &sbuf[wave][0][0][0];
    for (int idx = lane; idx < 2 * 16 * LDSS; idx += 32) p[idx] = (_Float16)0.f;
  }

  v16h a[3];
#pragma unroll
  for (int ks = 0; ks < 3; ++ks)
    a[ks] = load_a_frag(Z + rowBase * KT, KT, ks, lane);

  int pb = 0;
#pragma unroll
  for (int layer = 0; layer < 3; ++layer) {
    const _Float16* W = (layer == 0) ? W0 : (layer == 1) ? W1 : W2;
    _Float16 (*dst)[LDSS] = sbuf[wave][pb];
#pragma unroll
    for (int nt = 0; nt < 5; ++nt) {
      v8f acc = {};
#pragma unroll
      for (int ks = 0; ks < 3; ++ks)
        acc = wmma16(a[ks], load_b_frag(W + nt * 16 * KT, KT, ks, lane), acc);
#pragma unroll
      for (int j = 0; j < 8; ++j)
        dst[half * 8 + j][nt * 16 + ln] = (_Float16)fast_tanh(acc[j]);
    }
    if (lane < 16) dst[lane][66] = (_Float16)1.f;   // bias channel for next layer
#pragma unroll
    for (int ks = 0; ks < 3; ++ks)
      a[ks] = load_a_frag(&dst[0][0], LDSS, ks, lane);
    pb ^= 1;
  }

  // layer 3: (B,96) @ W3(96x16) -> 8 outputs; emit h-tilde channels
  v8f acc = {};
#pragma unroll
  for (int ks = 0; ks < 3; ++ks)
    acc = wmma16(a[ks], load_b_frag(W3, KT, ks, lane), acc);
#pragma unroll
  for (int j = 0; j < 8; ++j) {
    const float v = (ln < 8) ? fast_tanh(acc[j]) : ((ln == 8) ? 1.f : 0.f);
    HT[(rowBase + half * 8 + j) * 16 + ln] = v;
  }
}

// ---------------------------------------------------------------------------
// Hypernetwork apply: per wave = 16 rows.
//  GEMM1: X(16x160) x A1 -> per-m 16x16 tile of S; hid[.,m] = tanh(sum_kk ht*S)
//  GEMM2: hid(16x96) x A2 -> per-o tile of T; out[.,o] = sum_kk ht*(T + b2w)
// Reduction over the 16 packed k-channels = ds_swizzle XOR butterfly in each
// 16-lane half.
// ---------------------------------------------------------------------------
__global__ __launch_bounds__(256) void hyper_kernel(
    const _Float16* __restrict__ XH, const float* __restrict__ HT,
    const _Float16* __restrict__ A1, const _Float16* __restrict__ A2,
    const float* __restrict__ B2W, float* __restrict__ out) {
  __shared__ __align__(32) _Float16 hidbuf[8][16][LDSS];
  const int lane = threadIdx.x & 31;
  const int wave = threadIdx.x >> 5;
  const int half = lane >> 4, ln = lane & 15;
  const int rowBase = (blockIdx.x * 8 + wave) * 16;

  // h-tilde channel for this lane's column, for each of its 8 output rows
  float ht[8];
#pragma unroll
  for (int j = 0; j < 8; ++j)
    ht[j] = HT[(rowBase + half * 8 + j) * 16 + ln];

  // resident X fragments (5 k-steps)
  v16h xa[5];
#pragma unroll
  for (int ks = 0; ks < 5; ++ks)
    xa[ks] = load_a_frag(XH + rowBase * KH, KH, ks, lane);

  // clean hid tile (K pad columns must be zero, not LDS garbage)
  _Float16 (*hb)[LDSS] = hidbuf[wave];
  {
    _Float16* p = &hb[0][0];
    for (int idx = lane; idx < 16 * LDSS; idx += 32) p[idx] = (_Float16)0.f;
  }

  // ---- GEMM1: one 16x16 tile per hidden unit m ----
  for (int m = 0; m < 65; ++m) {
    v8f acc = {};
    const _Float16* bbase = A1 + (m * 16) * KH;
#pragma unroll
    for (int ks = 0; ks < 5; ++ks)
      acc = wmma16(xa[ks], load_b_frag(bbase, KH, ks, lane), acc);
    float red[8];
#pragma unroll
    for (int j = 0; j < 8; ++j) red[j] = acc[j] * ht[j];
    butterfly16(red);
    if (ln == 0) {
#pragma unroll
      for (int j = 0; j < 8; ++j)
        hb[half * 8 + j][m] = (_Float16)fast_tanh(red[j]);
    }
  }

  // hid fragments (3 k-steps over K2=96)
  v16h ha[3];
#pragma unroll
  for (int ks = 0; ks < 3; ++ks)
    ha[ks] = load_a_frag(&hb[0][0], LDSS, ks, lane);

  // ---- GEMM2: one 16x16 tile per output unit o ----
  for (int o = 0; o < 64; ++o) {
    v8f acc = {};
    const _Float16* bbase = A2 + (o * 16) * K2;
#pragma unroll
    for (int ks = 0; ks < 3; ++ks)
      acc = wmma16(ha[ks], load_b_frag(bbase, K2, ks, lane), acc);
    const float bw = B2W[o * 16 + ln];
    float red[8];
#pragma unroll
    for (int j = 0; j < 8; ++j) red[j] = (acc[j] + bw) * ht[j];
    butterfly16(red);
    if (ln == 0) {
#pragma unroll
      for (int j = 0; j < 8; ++j)
        out[(rowBase + half * 8 + j) * DIMC + o] = red[j];
    }
  }
}

// ---------------------------------------------------------------------------
// Host launcher
// ---------------------------------------------------------------------------
extern "C" void kernel_launch(void* const* d_in, const int* in_sizes, int n_in,
                              void* d_out, int out_size, void* d_ws, size_t ws_size,
                              hipStream_t stream) {
  const float* t      = (const float*)d_in[0];
  const float* y      = (const float*)d_in[1];
  const float* freqs  = (const float*)d_in[2];
  const float* hw0    = (const float*)d_in[3];
  const float* hb0    = (const float*)d_in[4];
  const float* hw1    = (const float*)d_in[5];
  const float* hb1    = (const float*)d_in[6];
  const float* hw2    = (const float*)d_in[7];
  const float* hb2    = (const float*)d_in[8];
  const float* hw3    = (const float*)d_in[9];
  const float* hb3    = (const float*)d_in[10];
  const float* head_w = (const float*)d_in[11];
  const float* head_b = (const float*)d_in[12];

  char* wsb = (char*)d_ws;
  size_t off = 0;
  auto carve = [&](size_t bytes) -> void* {
    void* p = wsb + off;
    off += (bytes + 255) & ~(size_t)255;
    return p;
  };
  _Float16* Z   = (_Float16*)carve((size_t)BN * KT * 2);   // 3.0 MB
  _Float16* XH  = (_Float16*)carve((size_t)BN * KH * 2);   // 5.0 MB
  float*    HT  = (float*)   carve((size_t)BN * 16 * 4);   // 1.0 MB
  _Float16* W0  = (_Float16*)carve((size_t)NT * KT * 2);
  _Float16* W1  = (_Float16*)carve((size_t)NT * KT * 2);
  _Float16* W2  = (_Float16*)carve((size_t)NT * KT * 2);
  _Float16* W3  = (_Float16*)carve((size_t)16 * KT * 2);
  _Float16* A1  = (_Float16*)carve((size_t)N1 * KH * 2);   // 333 KB (L2-resident)
  _Float16* A2  = (_Float16*)carve((size_t)N2 * K2 * 2);   // 196 KB (L2-resident)
  float*    B2W = (float*)   carve((size_t)1024 * 4);
  (void)ws_size; (void)in_sizes; (void)n_in; (void)out_size;

  prep_inputs_kernel<<<(BN * KH + 255) / 256, 256, 0, stream>>>(t, y, freqs, Z, XH);
  build_trunk_weights_kernel<<<(NT * KT + 255) / 256, 256, 0, stream>>>(
      hw0, hb0, hw1, hb1, hw2, hb2, hw3, hb3, W0, W1, W2, W3);
  build_head_weights_kernel<<<(N1 * KH + 255) / 256, 256, 0, stream>>>(
      head_w, head_b, A1, A2, B2W);
  trunk_kernel<<<BN / 128, 256, 0, stream>>>(Z, W0, W1, W2, W3, HT);
  hyper_kernel<<<BN / 128, 256, 0, stream>>>(XH, HT, A1, A2, B2W, (float*)d_out);
}
